// MultiheadAttention_79620103733666
// MI455X (gfx1250) — matrix operand order repaired and checked by reading, not compiled, not run on hardware
//
#include <hip/hip_runtime.h>

// MI455X / gfx1250: wave32, WMMA f32_16x16x32_f16.
// Pipeline: [f32->f16 prep / transposes] -> flash attention (WMMA) -> WMMA GEMM.
// Roofline: ~43 GFLOP vs ~80 MB -> matrix-pipe bound; all O(S^2) work runs on
// v_wmma; all f32->f16 conversion & layout shuffling is hoisted to O(S) preps.

typedef __attribute__((ext_vector_type(16))) _Float16 v16h;
typedef __attribute__((ext_vector_type(8)))  float    v8f;

constexpr int Bb = 2, Hh = 16, Ss = 2048, Dd = 64, DM = 1024;

union AFrag {
    v16h     v;
    _Float16 h[16];
    uint4    q[2];
};

// Immediate-form xor-shuffle (group-of-32 swizzle: and=0x1f, xor=M).
template <int M>
__device__ inline float swz_xor(float x) {
    return __int_as_float(
        __builtin_amdgcn_ds_swizzle(__float_as_int(x), (M << 10) | 0x1F));
}

// ---------------------------------------------------------------------------
// Prep 1: elementwise f32 -> f16 with scale (vectorized float4 -> 4 halves).
// ---------------------------------------------------------------------------
__global__ void __launch_bounds__(256)
cvt_half(const float* __restrict__ src, _Float16* __restrict__ dst,
         float scale)
{
    size_t i = ((size_t)blockIdx.x * blockDim.x + threadIdx.x) * 4;
    float4 f = *(const float4*)(src + i);
    union { _Float16 h[4]; uint2 u; } o;
    o.h[0] = (_Float16)(f.x * scale);
    o.h[1] = (_Float16)(f.y * scale);
    o.h[2] = (_Float16)(f.z * scale);
    o.h[3] = (_Float16)(f.w * scale);
    *(uint2*)(dst + i) = o.u;
}

// ---------------------------------------------------------------------------
// Prep 2: batched 16x16 tiled transpose + f32 -> f16.
//   src[b][r][c] (f32) -> dst[b][c][r] (f16).  block (16,16), grid (C/16,R/16,B)
// ---------------------------------------------------------------------------
__global__ void __launch_bounds__(256)
transpose_cvt(const float* __restrict__ src, _Float16* __restrict__ dst,
              int R, int C)
{
    __shared__ float t[16][17];
    const float* s = src + (size_t)blockIdx.z * R * C;
    _Float16*    d = dst + (size_t)blockIdx.z * R * C;
    int c0 = blockIdx.x * 16, r0 = blockIdx.y * 16;
    t[threadIdx.y][threadIdx.x] =
        s[(size_t)(r0 + threadIdx.y) * C + c0 + threadIdx.x];
    __syncthreads();
    d[(size_t)(c0 + threadIdx.y) * R + r0 + threadIdx.x] =
        (_Float16)t[threadIdx.x][threadIdx.y];
}

// ---------------------------------------------------------------------------
// Kernel 1: flash attention, all-f16 inputs. One wave per 16-query tile,
// 4 waves/block, grid = B*H*(S/64) = 1024. Output fused into (B,S,H*D) f16.
//   Qh: (B,H,S,D) f16 pre-scaled by 1/sqrt(D);  Kh: (B,H,S,D) f16;
//   Vt: (B,H,D,S) f16 (transposed).
// ---------------------------------------------------------------------------
__global__ void __launch_bounds__(128)
attn_fwd(const _Float16* __restrict__ Qh, const _Float16* __restrict__ Kh,
         const _Float16* __restrict__ Vt, _Float16* __restrict__ X)
{
    // Per-wave P-staging tile: 16 rows x 32 cols f16, padded to 40 halves/row.
    __shared__ _Float16 lds[4][16][40];

    const int lane = threadIdx.x & 31;
    const int wave = threadIdx.x >> 5;
    const int n    = lane & 15;   // N index within tile
    const int g    = lane >> 4;   // half-wave group

    int idx    = blockIdx.x;
    int qchunk = idx & 31;        // S/64 chunks
    int bh     = idx >> 5;
    int h      = bh & 15;
    int b      = bh >> 4;
    int q0     = qchunk * 64 + wave * 16;

    const size_t plane = (size_t)(b * Hh + h) * Ss * Dd;
    const _Float16* Qp = Qh + plane;
    const _Float16* Kp = Kh + plane;
    const _Float16* Vp = Vt + plane;   // [d][s]

    // Q A-fragments (two K=32 halves of D=64), pure b128 loads.
    // A layout: lane m = lane%16; half t -> k = (t/8)*16 + g*8 + t%8
    AFrag qa[2];
    {
        const _Float16* qrow = Qp + (size_t)(q0 + n) * Dd;
        #pragma unroll
        for (int kk = 0; kk < 2; ++kk) {
            qa[kk].q[0] = *(const uint4*)(qrow + kk * 32 + g * 8);
            qa[kk].q[1] = *(const uint4*)(qrow + kk * 32 + 16 + g * 8);
        }
    }

    // Softmax state. Row m = r + 8*g. mrun uniform over half-wave lanes;
    // lrun is a PER-LANE partial (this lane's two columns) -> reduced once at end.
    float mrun[8], lrun[8];
    v8f acc[4];
    #pragma unroll
    for (int r = 0; r < 8; ++r) { mrun[r] = -1e30f; lrun[r] = 0.0f; }
    #pragma unroll
    for (int t = 0; t < 4; ++t) acc[t] = (v8f){};

    for (int kt = 0; kt < Ss / 32; ++kt) {
        const int k0 = kt * 32;

        // K^T B-fragments: lane col n; half t -> k = (t/8)*16 + g*8 + t%8.
        AFrag kb[2][2];
        #pragma unroll
        for (int j = 0; j < 2; ++j) {
            const _Float16* krow = Kp + (size_t)(k0 + j * 16 + n) * Dd;
            #pragma unroll
            for (int kk = 0; kk < 2; ++kk) {
                kb[j][kk].q[0] = *(const uint4*)(krow + kk * 32 + g * 8);
                kb[j][kk].q[1] = *(const uint4*)(krow + kk * 32 + 16 + g * 8);
            }
        }

        // V B-fragments from transposed layout: two b128 loads each.
        AFrag vb[4];
        #pragma unroll
        for (int t = 0; t < 4; ++t) {
            const _Float16* vrow = Vp + (size_t)(t * 16 + n) * Ss + k0 + g * 8;
            vb[t].q[0] = *(const uint4*)(vrow);
            vb[t].q[1] = *(const uint4*)(vrow + 16);
        }

        // Scores: 16x32 tile as two 16x16 C tiles.
        v8f sc[2];
        #pragma unroll
        for (int j = 0; j < 2; ++j) {
            v8f z = (v8f){};
            z = __builtin_amdgcn_wmma_f32_16x16x32_f16(
                    false, qa[0].v, false, kb[j][0].v, (short)0, z, false, false);
            z = __builtin_amdgcn_wmma_f32_16x16x32_f16(
                    false, qa[1].v, false, kb[j][1].v, (short)0, z, false, false);
            sc[j] = z;
        }

        // Row max (ds_swizzle butterfly over the 16 lanes of each half-wave).
        float newm[8], alpha[8];
        #pragma unroll
        for (int r = 0; r < 8; ++r) {
            float v = fmaxf(sc[0][r], sc[1][r]);
            v = fmaxf(v, swz_xor<1>(v));
            v = fmaxf(v, swz_xor<2>(v));
            v = fmaxf(v, swz_xor<4>(v));
            v = fmaxf(v, swz_xor<8>(v));
            float nm = fmaxf(mrun[r], v);
            alpha[r] = __expf(mrun[r] - nm);
            newm[r]  = nm;
        }

        __syncthreads();  // LDS slice: previous reads done before new writes

        // P = exp(S - m) -> LDS (f16); per-lane partial row sums (no shuffle).
        #pragma unroll
        for (int r = 0; r < 8; ++r) {
            float p0 = __expf(sc[0][r] - newm[r]);
            float p1 = __expf(sc[1][r] - newm[r]);
            lds[wave][r + g * 8][n]      = (_Float16)p0;
            lds[wave][r + g * 8][16 + n] = (_Float16)p1;
            lrun[r] = lrun[r] * alpha[r] + (p0 + p1);
            mrun[r] = newm[r];
        }

        // Rescale O accumulators (alpha uniform across half-wave lanes).
        #pragma unroll
        for (int t = 0; t < 4; ++t)
            #pragma unroll
            for (int r = 0; r < 8; ++r)
                acc[t][r] *= alpha[r];

        __syncthreads();

        // Re-load P as interleaved A-fragment.
        AFrag pa;
        {
            const char* row = (const char*)&lds[wave][n][0];
            pa.q[0] = *(const uint4*)(row + g * 16);
            pa.q[1] = *(const uint4*)(row + 32 + g * 16);
        }

        // O += P @ V
        #pragma unroll
        for (int t = 0; t < 4; ++t)
            acc[t] = __builtin_amdgcn_wmma_f32_16x16x32_f16(
                         false, pa.v, false, vb[t].v, (short)0, acc[t],
                         false, false);
    }

    // Final cross-lane sum of the per-lane partial l (once, not per tile).
    float lsum[8];
    #pragma unroll
    for (int r = 0; r < 8; ++r) {
        float s = lrun[r];
        s += swz_xor<1>(s);
        s += swz_xor<2>(s);
        s += swz_xor<4>(s);
        s += swz_xor<8>(s);
        lsum[r] = s;
    }

    // Normalize, store fused (B, S, H*D) f16.
    #pragma unroll
    for (int t = 0; t < 4; ++t)
        #pragma unroll
        for (int r = 0; r < 8; ++r) {
            int q = q0 + r + g * 8;
            float o = acc[t][r] / lsum[r];
            X[(size_t)(b * Ss + q) * DM + h * Dd + t * 16 + n] = (_Float16)o;
        }
}

// ---------------------------------------------------------------------------
// Kernel 2: out = X @ W + b with W pre-transposed to f16 (Wt[n][k]).
// grid = (4096/16, 1024/64), block = 128 (wave w owns N-tile w).
// ---------------------------------------------------------------------------
__global__ void __launch_bounds__(128)
proj_gemm(const _Float16* __restrict__ X, const _Float16* __restrict__ Wt,
          const float* __restrict__ bias, float* __restrict__ out)
{
    const int lane = threadIdx.x & 31;
    const int wave = threadIdx.x >> 5;
    const int n    = lane & 15;
    const int g    = lane >> 4;

    const int m0 = blockIdx.x * 16;
    const int n0 = blockIdx.y * 64 + wave * 16;

    v8f c = (v8f){};
    const _Float16* xrow = X + (size_t)(m0 + n) * DM;
    const _Float16* wrow = Wt + (size_t)(n0 + n) * DM;

    for (int kb = 0; kb < DM / 32; ++kb) {
        const int kbase = kb * 32;

        AFrag a;   // A interleaved layout: two b128 loads
        a.q[0] = *(const uint4*)(xrow + kbase + g * 8);
        a.q[1] = *(const uint4*)(xrow + kbase + 16 + g * 8);

        AFrag bw;  // B interleaved layout: k = kbase + (t/8)*16 + g*8 + t%8
        bw.q[0] = *(const uint4*)(wrow + kbase + g * 8);
        bw.q[1] = *(const uint4*)(wrow + kbase + 16 + g * 8);

        c = __builtin_amdgcn_wmma_f32_16x16x32_f16(
                false, a.v, false, bw.v, (short)0, c, false, false);
    }

    const float bn = bias[n0 + n];
    #pragma unroll
    for (int r = 0; r < 8; ++r) {
        int m = m0 + r + g * 8;
        out[(size_t)m * DM + n0 + n] = c[r] + bn;
    }
}

// ---------------------------------------------------------------------------
extern "C" void kernel_launch(void* const* d_in, const int* in_sizes, int n_in,
                              void* d_out, int out_size, void* d_ws,
                              size_t ws_size, hipStream_t stream)
{
    const float* Q    = (const float*)d_in[0];
    const float* K    = (const float*)d_in[1];
    const float* V    = (const float*)d_in[2];
    const float* W    = (const float*)d_in[3];
    const float* bias = (const float*)d_in[4];
    float*       out  = (float*)d_out;

    constexpr size_t NQKV = (size_t)Bb * Hh * Ss * Dd;  // 4,194,304
    constexpr size_t NX   = (size_t)Bb * Ss * DM;       // 4,194,304
    constexpr size_t NW   = (size_t)DM * DM;            // 1,048,576

    _Float16* ws = (_Float16*)d_ws;                     // 34 MB total
    _Float16* Qh = ws;
    _Float16* Kh = Qh + NQKV;
    _Float16* Vt = Kh + NQKV;
    _Float16* X  = Vt + NQKV;
    _Float16* Wt = X + NX;

    // Preps (bandwidth-trivial: ~60 MB total @ 23.3 TB/s).
    cvt_half<<<dim3(NQKV / (256 * 4)), dim3(256), 0, stream>>>(Q, Qh, 0.125f);
    cvt_half<<<dim3(NQKV / (256 * 4)), dim3(256), 0, stream>>>(K, Kh, 1.0f);
    transpose_cvt<<<dim3(Dd / 16, Ss / 16, Bb * Hh), dim3(16, 16), 0, stream>>>(
        V, Vt, Ss, Dd);
    transpose_cvt<<<dim3(DM / 16, DM / 16, 1), dim3(16, 16), 0, stream>>>(
        W, Wt, DM, DM);

    attn_fwd<<<dim3(Bb * Hh * (Ss / 64)), dim3(128), 0, stream>>>(
        Qh, Kh, Vt, X);
    proj_gemm<<<dim3((Bb * Ss) / 16, DM / 64), dim3(128), 0, stream>>>(
        X, Wt, bias, out);
}
